// MultiHeadAttention_87857851007775
// MI455X (gfx1250) — matrix operand order repaired and checked by reading, not compiled, not run on hardware
//
#include <hip/hip_runtime.h>

#define BB 32
#define TT 512
#define EE 768
#define HH 12
#define DD 64
#define BT (BB * TT) /* 16384 */

typedef __attribute__((ext_vector_type(16))) __bf16        v16bf;
typedef __attribute__((ext_vector_type(8)))  float         v8f;
typedef __attribute__((ext_vector_type(4)))  unsigned int  u32x4;
typedef __attribute__((ext_vector_type(8)))  unsigned int  u32x8;
typedef __attribute__((ext_vector_type(2)))  unsigned int  u32x2;
typedef __attribute__((ext_vector_type(4)))  float         f32x4;

union Frag {
  v16bf v;
  u32x4 q[2];
};

__device__ __forceinline__ unsigned short f32_to_bf16(float f) {
  unsigned int u = __builtin_bit_cast(unsigned int, f);
  u += 0x7FFFu + ((u >> 16) & 1u); // round-to-nearest-even
  return (unsigned short)(u >> 16);
}

__device__ __forceinline__ v8f wmma_bf16(const Frag& a, const Frag& b, v8f c) {
  return __builtin_amdgcn_wmma_f32_16x16x32_bf16(false, a.v, false, b.v,
                                                 (short)0, c, false, false);
}

// per-lane async copy: 16B global -> LDS, tracked by ASYNCcnt
__device__ __forceinline__ void async_cp16(unsigned short* lds_ptr,
                                           const unsigned short* gptr) {
  unsigned lds = (unsigned)(unsigned long long)lds_ptr; // low 32 = LDS offset
  unsigned long long g = (unsigned long long)gptr;
  asm volatile("global_load_async_to_lds_b128 %0, %1, off"
               :: "v"(lds), "v"(g) : "memory");
}
__device__ __forceinline__ void wait_async0() {
  asm volatile("s_wait_asynccnt 0" ::: "memory");
}

// TDM 2D tile load (bf16 elements) with hardware row padding.
// D# layout per CDNA5 ISA.
__device__ __forceinline__ void tdm_load_2d(unsigned short* lds_ptr,
                                            const unsigned short* gptr,
                                            unsigned tile_w, unsigned tile_h,
                                            unsigned tensor_w, unsigned tensor_h,
                                            unsigned row_stride,
                                            unsigned pad_interval_code,
                                            unsigned pad_amount_code,
                                            bool pad_en) {
  unsigned long long ga = (unsigned long long)gptr;
  unsigned lds = (unsigned)(unsigned long long)lds_ptr;
  u32x4 g0;
  g0[0] = 1u;                                               // count=1 (user)
  g0[1] = lds;                                              // lds_addr
  g0[2] = (unsigned)(ga & 0xFFFFFFFFu);                     // global_addr lo
  g0[3] = (unsigned)((ga >> 32) & 0x1FFFFFFu) | (2u << 30); // ga[56:32] | type=2
  u32x8 g1;
  g1[0] = (1u << 16)                                        // data_size = 2B
        | (pad_en ? (1u << 20) : 0u)
        | (pad_interval_code << 22) | (pad_amount_code << 25);
  g1[1] = (tensor_w & 0xFFFFu) << 16;                       // tensor_dim0[15:0]
  g1[2] = (tensor_w >> 16) | ((tensor_h & 0xFFFFu) << 16);
  g1[3] = (tensor_h >> 16) | (tile_w << 16);                // tile_dim0
  g1[4] = tile_h & 0xFFFFu;                                 // tile_dim1
  g1[5] = row_stride;                                       // stride0[31:0]
  g1[6] = 0;
  g1[7] = 0;
  asm volatile("tensor_load_to_lds %0, %1" :: "s"(g0), "s"(g1) : "memory");
}

// ---------------------------------------------------------------- cvt f32->bf16
__global__ void cvt_f32_bf16(const float* __restrict__ src,
                             unsigned short* __restrict__ dst, int n) {
  int i = (blockIdx.x * blockDim.x + threadIdx.x) * 4;
  if (i + 3 < n) {
    f32x4 f = *(const f32x4*)(src + i);
    unsigned int u0 = (unsigned int)f32_to_bf16(f.x) |
                      ((unsigned int)f32_to_bf16(f.y) << 16);
    unsigned int u1 = (unsigned int)f32_to_bf16(f.z) |
                      ((unsigned int)f32_to_bf16(f.w) << 16);
    u32x2 p; p.x = u0; p.y = u1;
    *(u32x2*)(dst + i) = p;
  } else if (i < n) {
    for (int j = i; j < n; ++j) dst[j] = f32_to_bf16(src[j]);
  }
}

// ---------------------------------------------------------------- bf16 GEMM
// C[m,n] = sum_k A[m,k] * W[n,k] + bias[n]
// mode 0: bf16 out, [B,H,T,D]   mode 2: bf16 out, [B,H,D,T] (V^T)   mode 1: f32
// 256 thr = 8 waves (4M x 2N); block tile 128M x 128N; per-wave 32M x 64N;
// K step 32; async double-buffered LDS staging (40 KB).
__global__ __launch_bounds__(256) void gemm_bf16(
    const unsigned short* __restrict__ A,  // [16384,768] bf16 row-major
    const unsigned short* __restrict__ W,  // [768,768] bf16 row-major [out,in]
    const float* __restrict__ bias,        // [768]
    unsigned short* __restrict__ outb, float* __restrict__ outf, int mode) {
  __shared__ unsigned short As[2][128 * 40];   // padded stride 40
  __shared__ unsigned short Bs[2][128 * 40];

  const int tid  = threadIdx.x;
  const int lane = tid & 31;
  const int w    = tid >> 5;
  const int wm   = w & 3;            // 4 waves over M
  const int wn   = w >> 2;           // 2 waves over N
  const int ln16 = lane & 15;
  const int hi   = lane >> 4;
  const int aofs = hi ? 8 : 0;
  const int m0   = blockIdx.x * 128;
  const int n0   = blockIdx.y * 128;

  v8f acc[2][4];
  for (int mt = 0; mt < 2; ++mt)
    for (int nt = 0; nt < 4; ++nt)
      for (int r = 0; r < 8; ++r) acc[mt][nt][r] = 0.0f;

  auto stage = [&](int buf, int k0) {
    // A: 128 rows x 32 k = 512 x 16B chunks; B: same
    for (int it = 0; it < 2; ++it) {
      int c = it * 256 + tid, row = c >> 2, cc = c & 3;
      async_cp16(&As[buf][row * 40 + cc * 8],
                 A + (size_t)(m0 + row) * EE + k0 + cc * 8);
    }
    for (int it = 0; it < 2; ++it) {
      int c = it * 256 + tid, row = c >> 2, cc = c & 3;
      async_cp16(&Bs[buf][row * 40 + cc * 8],
                 W + (size_t)(n0 + row) * EE + k0 + cc * 8);
    }
  };

  stage(0, 0);
  wait_async0();
  __syncthreads();

#pragma unroll 2
  for (int it = 0; it < EE / 32; ++it) {
    const int cur = it & 1;
    if (it + 1 < EE / 32) stage(cur ^ 1, (it + 1) * 32); // overlap with WMMA

    Frag a[2];
#pragma unroll
    for (int mt = 0; mt < 2; ++mt) {
      const unsigned short* ar = &As[cur][(wm * 32 + mt * 16 + ln16) * 40 + aofs];
      a[mt].q[0] = *(const u32x4*)ar;
      a[mt].q[1] = *(const u32x4*)(ar + 16);
    }
#pragma unroll
    for (int nt = 0; nt < 4; ++nt) {
      Frag bfr;
      const unsigned short* br =
          &Bs[cur][(wn * 64 + nt * 16 + ln16) * 40 + hi * 8];
      bfr.q[0] = *(const u32x4*)br;
      bfr.q[1] = *(const u32x4*)(br + 16);
#pragma unroll
      for (int mt = 0; mt < 2; ++mt)
        acc[mt][nt] = wmma_bf16(a[mt], bfr, acc[mt][nt]);
    }
    wait_async0();
    __syncthreads();
  }

#pragma unroll
  for (int mt = 0; mt < 2; ++mt) {
#pragma unroll
    for (int nt = 0; nt < 4; ++nt) {
#pragma unroll
      for (int r = 0; r < 8; ++r) {
        int m = m0 + wm * 32 + mt * 16 + r + 8 * hi;
        int n = n0 + wn * 64 + nt * 16 + ln16;
        float v = acc[mt][nt][r] + bias[n];
        int b = m >> 9, t = m & 511, h = n >> 6, d = n & 63;
        if (mode == 0) {
          outb[(((size_t)(b * HH + h) * TT) + t) * DD + d] = f32_to_bf16(v);
        } else if (mode == 2) { // V^T: [B,H,D,T]
          outb[(((size_t)(b * HH + h) * DD) + d) * TT + t] = f32_to_bf16(v);
        } else {
          outf[(size_t)m * EE + n] = v;
        }
      }
    }
  }
}

// ---------------------------------------------------------------- fused attention
// grid: (T/64, B*H); block: 128 thr = 4 waves, each wave owns 16 q rows.
// K and V^T chunks staged by the Tensor Data Mover with hardware row padding.
__global__ __launch_bounds__(128) void attn_kernel(
    const unsigned short* __restrict__ Q,   // [B,H,T,D] bf16
    const unsigned short* __restrict__ K,   // [B,H,T,D] bf16
    const unsigned short* __restrict__ Vt,  // [B,H,D,T] bf16 (pre-transposed)
    const int* __restrict__ am,             // [B,T]
    unsigned short* __restrict__ ctx) {     // [B,T,E] bf16 merged
  __shared__ unsigned short Ks[128 * 72];    // [t][d], TDM-padded stride 72
  __shared__ unsigned short Vts[64 * 136];   // [d][t], TDM-padded stride 136
  __shared__ unsigned short Ps[4 * 16 * 48]; // per-wave P tile 16x32, pad 48

  const int bh = blockIdx.y;
  const int b = bh / HH, h = bh % HH;
  const size_t hb = (size_t)bh * TT * DD;
  const int tid  = threadIdx.x;
  const int lane = tid & 31;
  const int w    = tid >> 5;
  const int ln16 = lane & 15;
  const int hi   = lane >> 4;
  const int aofs = hi ? 8 : 0;
  const int q0   = blockIdx.x * 64 + w * 16;

  // Q A-fragments (d = 0..31, 32..63) held in registers for whole kernel
  Frag qa[2];
  {
    const unsigned short* qr = Q + hb + (size_t)(q0 + ln16) * DD;
#pragma unroll
    for (int kkf = 0; kkf < 2; ++kkf) {
      qa[kkf].q[0] = *(const u32x4*)(qr + kkf * 32 + aofs);
      qa[kkf].q[1] = *(const u32x4*)(qr + kkf * 32 + aofs + 16);
    }
  }

  float m_run[8], l_run[8];
  v8f acc[4];
  for (int r = 0; r < 8; ++r) { m_run[r] = -3.0e38f; l_run[r] = 0.0f; }
  for (int dt = 0; dt < 4; ++dt)
    for (int r = 0; r < 8; ++r) acc[dt][r] = 0.0f;

  for (int tc = 0; tc < TT; tc += 128) {
    if (tc + 128 < TT) { // warm L2 for the next chunk
      __builtin_prefetch(K + hb + (size_t)(tc + 128) * DD + tid * 16, 0, 1);
      __builtin_prefetch(Vt + hb + tc + 128 + tid * 256, 0, 1);
    }
    if (w == 0) { // one wave drives the TDM; EXEC-independent wave op
      // K chunk: 128 rows x 64 elems, stride 64; pad 32dw rows by 4dw -> 72
      tdm_load_2d(Ks, K + hb + (size_t)tc * DD, 64, 128, 64, 128, 64,
                  /*32dw*/ 4, /*4dw*/ 3, true);
      // Vt chunk: 64 rows x 128 elems, stride 512; pad 64dw rows by 4dw -> 136
      tdm_load_2d(Vts, Vt + hb + tc, 128, 64, 128, 64, 512,
                  /*64dw*/ 5, /*4dw*/ 3, true);
      __builtin_amdgcn_s_wait_tensorcnt(0);
    }
    __syncthreads();

#pragma unroll
    for (int t0 = 0; t0 < 128; t0 += 32) {
      // scores for 32 key columns: two 16x16 C tiles
      v8f sc0, sc1;
      for (int r = 0; r < 8; ++r) { sc0[r] = 0.0f; sc1[r] = 0.0f; }
#pragma unroll
      for (int kkf = 0; kkf < 2; ++kkf) {
        Frag kb0, kb1;
        const unsigned short* k0p = &Ks[(t0 + ln16) * 72 + kkf * 32 + hi * 8];
        const unsigned short* k1p = &Ks[(t0 + 16 + ln16) * 72 + kkf * 32 + hi * 8];
        kb0.q[0] = *(const u32x4*)k0p; kb0.q[1] = *(const u32x4*)(k0p + 16);
        kb1.q[0] = *(const u32x4*)k1p; kb1.q[1] = *(const u32x4*)(k1p + 16);
        sc0 = wmma_bf16(qa[kkf], kb0, sc0);
        sc1 = wmma_bf16(qa[kkf], kb1, sc1);
      }
      // scale + mask (each lane owns one key column per tile)
      int tg0 = tc + t0 + ln16;
      bool z0 = (am[b * TT + tg0] == 0);
      bool z1 = (am[b * TT + tg0 + 16] == 0);
#pragma unroll
      for (int r = 0; r < 8; ++r) {
        sc0[r] = z0 ? -3.0e38f : sc0[r] * 0.125f;
        sc1[r] = z1 ? -3.0e38f : sc1[r] * 0.125f;
      }
      // online softmax; row m = r + 8*hi lives in one 16-lane half-wave
      float p0[8], p1[8], corr[8];
#pragma unroll
      for (int r = 0; r < 8; ++r) {
        float a = fmaxf(sc0[r], sc1[r]);
        a = fmaxf(a, __shfl_xor(a, 1, 32));
        a = fmaxf(a, __shfl_xor(a, 2, 32));
        a = fmaxf(a, __shfl_xor(a, 4, 32));
        a = fmaxf(a, __shfl_xor(a, 8, 32));
        float mn = fmaxf(m_run[r], a);
        corr[r] = __expf(m_run[r] - mn);
        p0[r] = __expf(sc0[r] - mn);
        p1[r] = __expf(sc1[r] - mn);
        float rs = p0[r] + p1[r];
        rs += __shfl_xor(rs, 1, 32);
        rs += __shfl_xor(rs, 2, 32);
        rs += __shfl_xor(rs, 4, 32);
        rs += __shfl_xor(rs, 8, 32);
        l_run[r] = l_run[r] * corr[r] + rs;
        m_run[r] = mn;
      }
#pragma unroll
      for (int dt = 0; dt < 4; ++dt)
#pragma unroll
        for (int r = 0; r < 8; ++r) acc[dt][r] *= corr[r];

      // C-layout -> A-layout for P via per-wave LDS tile
      __syncthreads();
      unsigned short* pw = &Ps[w * 768];
#pragma unroll
      for (int r = 0; r < 8; ++r) {
        int m = r + 8 * hi;
        pw[m * 48 + ln16]      = f32_to_bf16(p0[r]);
        pw[m * 48 + 16 + ln16] = f32_to_bf16(p1[r]);
      }
      __syncthreads();
      Frag pa;
      const unsigned short* pr = &Ps[w * 768 + ln16 * 48 + aofs];
      pa.q[0] = *(const u32x4*)pr;
      pa.q[1] = *(const u32x4*)(pr + 16);

      // ctx += P * V   (B operand contiguous from V^T)
#pragma unroll
      for (int dt = 0; dt < 4; ++dt) {
        Frag vb;
        const unsigned short* vr = &Vts[(dt * 16 + ln16) * 136 + t0 + hi * 8];
        vb.q[0] = *(const u32x4*)vr;
        vb.q[1] = *(const u32x4*)(vr + 16);
        acc[dt] = wmma_bf16(pa, vb, acc[dt]);
      }
    }
    __syncthreads();
  }

  // normalize and write merged [B,T,E]
#pragma unroll
  for (int dt = 0; dt < 4; ++dt) {
#pragma unroll
    for (int r = 0; r < 8; ++r) {
      int tq = q0 + r + 8 * hi;
      int d = dt * 16 + ln16;
      float v = acc[dt][r] / fmaxf(l_run[r], 1e-20f);
      ctx[((size_t)b * TT + tq) * EE + h * DD + d] = f32_to_bf16(v);
    }
  }
}

// ---------------------------------------------------------------- launch
extern "C" void kernel_launch(void* const* d_in, const int* in_sizes, int n_in,
                              void* d_out, int out_size, void* d_ws,
                              size_t ws_size, hipStream_t stream) {
  const float* x  = (const float*)d_in[0];
  const int* mask = (const int*)d_in[1];
  const float* wq = (const float*)d_in[2];
  const float* bq = (const float*)d_in[3];
  const float* wk = (const float*)d_in[4];
  const float* bk = (const float*)d_in[5];
  const float* wv = (const float*)d_in[6];
  const float* bv = (const float*)d_in[7];
  const float* wo = (const float*)d_in[8];
  const float* bo = (const float*)d_in[9];
  float* out = (float*)d_out;

  char* ws = (char*)d_ws;
  size_t off = 0;
  auto alloc = [&](size_t bytes) {
    void* p = ws + off;
    off += (bytes + 255) & ~(size_t)255;
    return p;
  };
  const size_t SZ_X = (size_t)BT * EE * 2;
  const size_t SZ_W = (size_t)EE * EE * 2;
  unsigned short* xb   = (unsigned short*)alloc(SZ_X);
  unsigned short* wqb  = (unsigned short*)alloc(SZ_W);
  unsigned short* wkb  = (unsigned short*)alloc(SZ_W);
  unsigned short* wvb  = (unsigned short*)alloc(SZ_W);
  unsigned short* wob  = (unsigned short*)alloc(SZ_W);
  unsigned short* Qb   = (unsigned short*)alloc(SZ_X);
  unsigned short* Kb   = (unsigned short*)alloc(SZ_X);
  unsigned short* Vtb  = (unsigned short*)alloc(SZ_X);
  unsigned short* ctxb = (unsigned short*)alloc(SZ_X);

  {
    int n = BT * EE;
    cvt_f32_bf16<<<(n / 4 + 255) / 256, 256, 0, stream>>>(x, xb, n);
    int nw = EE * EE;
    int gw = (nw / 4 + 255) / 256;
    cvt_f32_bf16<<<gw, 256, 0, stream>>>(wq, wqb, nw);
    cvt_f32_bf16<<<gw, 256, 0, stream>>>(wk, wkb, nw);
    cvt_f32_bf16<<<gw, 256, 0, stream>>>(wv, wvb, nw);
    cvt_f32_bf16<<<gw, 256, 0, stream>>>(wo, wob, nw);
  }

  dim3 ggrid(BT / 128, EE / 128);
  gemm_bf16<<<ggrid, 256, 0, stream>>>(xb, wqb, bq, Qb, nullptr, 0);
  gemm_bf16<<<ggrid, 256, 0, stream>>>(xb, wkb, bk, Kb, nullptr, 0);
  gemm_bf16<<<ggrid, 256, 0, stream>>>(xb, wvb, bv, Vtb, nullptr, 2); // V^T

  dim3 agrid(TT / 64, BB * HH);
  attn_kernel<<<agrid, 128, 0, stream>>>(Qb, Kb, Vtb, mask, ctxb);

  gemm_bf16<<<ggrid, 256, 0, stream>>>(ctxb, wob, bo, nullptr, out, 1);
}
